// Encoder_66623532696154
// MI455X (gfx1250) — matrix operand order repaired and checked by reading, not compiled, not run on hardware
//
#include <hip/hip_runtime.h>
#include <hip/hip_bf16.h>

// Problem constants (match reference)
#define B_   64
#define T_   512
#define E_   512
#define H_   1024
#define K_   (E_ + H_)   // 1536 : fused K for [x_t, h] @ [W_ih ; W_hh]
#define N4H  (4 * H_)    // 4096

typedef __attribute__((ext_vector_type(16))) __bf16 v16bf;
typedef __attribute__((ext_vector_type(8)))  __bf16 v8bf;
typedef __attribute__((ext_vector_type(8)))  float  v8f;
typedef __attribute__((ext_vector_type(4)))  int    v4i;

// CDNA5 async global->LDS DMA (ASYNCcnt) if the toolchain exposes it.
// Probe round 2 confirmed the builtin exists with signature:
//   (addrspace(1) v4i* src, addrspace(3) v4i* dst, imm offset, imm cpol)
#if defined(__gfx1250__) && __has_builtin(__builtin_amdgcn_global_load_async_to_lds_b128)
#define USE_ASYNC_LDS 1
#else
#define USE_ASYNC_LDS 0
#endif

typedef __attribute__((address_space(1))) v4i* gv4i_ptr;
typedef __attribute__((address_space(3))) v4i* lv4i_ptr;

__device__ __forceinline__ void wait_asynccnt0() {
#if defined(__gfx1250__)
#if __has_builtin(__builtin_amdgcn_s_wait_asynccnt)
    __builtin_amdgcn_s_wait_asynccnt(0);
#else
    asm volatile("s_wait_asynccnt 0x0" ::: "memory");
#endif
#endif
}

__device__ __forceinline__ float sigmoidf_(float x) {
    return 1.0f / (1.0f + __expf(-x));
}

// ---------------------------------------------------------------------------
// Pack W_ih (E x 4H) and W_hh (H x 4H) fp32 into one bf16 matrix Wc (K_ x 4H).
// 12.6 MB bf16 -> stays resident in the 192 MB L2 across all 512 steps.
// ---------------------------------------------------------------------------
__global__ void pack_weights_kernel(const float* __restrict__ Wih,
                                    const float* __restrict__ Whh,
                                    __bf16* __restrict__ Wc) {
    const int total = K_ * N4H;
    for (int i = blockIdx.x * blockDim.x + threadIdx.x; i < total;
         i += gridDim.x * blockDim.x) {
        int k = i / N4H, n = i - k * N4H;
        float v = (k < E_) ? Wih[k * N4H + n] : Whh[(k - E_) * N4H + n];
        Wc[i] = (__bf16)v;
    }
}

// ---------------------------------------------------------------------------
// Embedding gather + fp32 -> bf16 convert: X[b][t][e] = bf16(emb[src[b,t]][e])
// ---------------------------------------------------------------------------
__global__ void embed_bf16_kernel(const int* __restrict__ src,
                                  const float* __restrict__ emb,
                                  __bf16* __restrict__ X) {
    const long total = (long)B_ * T_ * E_;
    for (long i = (long)blockIdx.x * blockDim.x + threadIdx.x; i < total;
         i += (long)gridDim.x * blockDim.x) {
        long bt = i / E_;
        int  e  = (int)(i - bt * E_);
        int  row = src[bt];
        X[i] = (__bf16)emb[(long)row * E_ + e];
    }
}

// ---------------------------------------------------------------------------
// Initialize step state: A0 = [x_0 | h=0] (bf16), c = 0 (fp32).
// Re-run every launch -> deterministic under graph replay.
// ---------------------------------------------------------------------------
__global__ void init_state_kernel(const __bf16* __restrict__ X,
                                  __bf16* __restrict__ A0,
                                  float* __restrict__ C) {
    int i = blockIdx.x * blockDim.x + threadIdx.x;
    if (i < B_ * K_) {
        int b = i / K_, col = i - b * K_;
        A0[i] = (col < E_) ? X[((long)b * T_) * E_ + col] : (__bf16)0.0f;
    }
    if (i < B_ * H_) C[i] = 0.0f;
}

// ---------------------------------------------------------------------------
// One LSTM timestep. grid = H_/16 = 64 workgroups, 256 threads = 8 waves.
// Workgroup g owns hidden columns [g*16, g*16+16) across ALL four gate blocks,
// so gates + c/h update are fully fused in-kernel.
//
// z-tile decomposition per WG: 64 batch rows x (4 gates x 16 cols)
//   = 16 WMMA 16x16 tiles; wave w computes tiles {w, w+8} (same gate column
//   block, row bases 16*(w>>2) and +32) so the B fragment is shared.
// K loop: 1536/32 = 48 chunks of v_wmma_f32_16x16x32_bf16, with double-
// buffered LDS: stage chunk k+1 (async DMA for A, transposing VALU path for B,
// global_prefetch_b8 for the chunk after) while computing chunk k.
// ---------------------------------------------------------------------------
__global__ __launch_bounds__(256)
void lstm_step_kernel(const __bf16* __restrict__ Ain,   // [B_][K_]  (x_t | h_{t-1})
                      __bf16* __restrict__ Aout,        // [B_][K_]  (x_{t+1} | h_t)
                      const __bf16* __restrict__ Wc,    // [K_][N4H]
                      const float* __restrict__ bias,   // [N4H]
                      const __bf16* __restrict__ X,     // [B_][T_][E_]
                      float* __restrict__ C,            // [B_][H_]
                      float* __restrict__ out,          // d_out: h_T then c_T
                      int t) {
    // Double-buffered staging; row strides padded to 40 elems (80B) to keep
    // 16B alignment of sub-row accesses and spread LDS banks.
    __shared__ __align__(16) __bf16 lds_a[2][B_][40];     // 64x32 K-chunks of A
    __shared__ __align__(16) __bf16 lds_b[2][4][16][40];  // [gate][n][k] transposed
    __shared__ float lds_z[B_][4 * 17];                   // z results, padded

    const int tid   = threadIdx.x;
    const int lane  = tid & 31;          // wave32
    const int wave  = tid >> 5;          // 0..7
    const int g16   = blockIdx.x * 16;   // hidden-column base for this WG
    const int gate  = wave & 3;          // 0=i 1=f 2=g 3=o
    const int col   = lane & 15;         // N within 16-wide tile
    const int khalf = lane >> 4;         // K-half select per ISA A/B layouts

    // Bias folded into accumulator init: in the C/D layout every lane's 8 VGPRs
    // hold one column, so a per-lane splat of bias[column] is exact.
    const int zcol = gate * H_ + g16 + col;
    const float bv = bias[zcol];
    v8f acc0, acc1;
#pragma unroll
    for (int j = 0; j < 8; ++j) { acc0[j] = bv; acc1[j] = bv; }

    const int rowA0 = (wave >> 2) * 16 + col;  // A-frag row, tile 0 (rows 0/16)
    const int rowA1 = rowA0 + 32;              // tile 1 (rows 32/48)

    // Staging work split (256 threads):
    const int ak_row = tid >> 2;               // 0..63  (A: 8 bf16 per thread)
    const int ak_seg = (tid & 3) * 8;          // 0,8,16,24
    const int bk_k   = tid >> 3;               // 0..31  (B: 8 bf16 per thread)
    const int bk_grp = (tid & 7) * 8;          // 0..56 within WG's 64 z-cols
    const int bk_gate = bk_grp >> 4;           // 0..3
    const int bk_n    = bk_grp & 15;           // 0 or 8
    const long bk_col = (long)bk_gate * H_ + g16 + bk_n;

    // Stage one 32-wide K chunk into LDS buffer `buf`.
    auto stage = [&](int kk, int buf) {
        // A chunk (64x32 bf16): contiguous 16B per lane.
#if USE_ASYNC_LDS
        __builtin_amdgcn_global_load_async_to_lds_b128(
            (gv4i_ptr)&Ain[ak_row * K_ + kk + ak_seg],
            (lv4i_ptr)&lds_a[buf][ak_row][ak_seg],
            0, 0);
#else
        {
            v8bf av = *(const v8bf*)&Ain[ak_row * K_ + kk + ak_seg];
            *(v8bf*)&lds_a[buf][ak_row][ak_seg] = av;
        }
#endif
        // B chunk transposed to [gate][n][k]; prefetch chunk after next
        // (weights are L2-resident; global_prefetch_b8).
        const __bf16* wp = &Wc[(long)(kk + bk_k) * N4H + bk_col];
        v8bf wv = *(const v8bf*)wp;
#pragma unroll
        for (int j = 0; j < 8; ++j) lds_b[buf][bk_gate][bk_n + j][bk_k] = wv[j];
        if (kk + 32 < K_) __builtin_prefetch(wp + (long)32 * N4H, 0, 1);
    };

    // Prologue: stage chunk 0.
    stage(0, 0);
#if USE_ASYNC_LDS
    wait_asynccnt0();
#endif
    __syncthreads();

    for (int kk = 0; kk < K_; kk += 32) {
        const int cur = (kk >> 5) & 1;
        // Stage next chunk into the other buffer (overlaps with WMMA below).
        if (kk + 32 < K_) stage(kk + 32, cur ^ 1);

        // B fragment (32x16 bf16): lane holds col=lane&15, K{kh*8..+8,16+kh*8..+8}.
        v16bf bfrag;
        {
            v8bf lo = *(const v8bf*)&lds_b[cur][gate][col][khalf * 8];
            v8bf hi = *(const v8bf*)&lds_b[cur][gate][col][16 + khalf * 8];
#pragma unroll
            for (int j = 0; j < 8; ++j) { bfrag[j] = lo[j]; bfrag[8 + j] = hi[j]; }
        }
        // A fragments (16x32 bf16): lane row = M, elems K{kh*8..+8,16+kh*8..+8}.
        v16bf afrag0, afrag1;
        {
            v8bf lo = *(const v8bf*)&lds_a[cur][rowA0][khalf * 8];
            v8bf hi = *(const v8bf*)&lds_a[cur][rowA0][16 + khalf * 8];
#pragma unroll
            for (int j = 0; j < 8; ++j) { afrag0[j] = lo[j]; afrag0[8 + j] = hi[j]; }
        }
        {
            v8bf lo = *(const v8bf*)&lds_a[cur][rowA1][khalf * 8];
            v8bf hi = *(const v8bf*)&lds_a[cur][rowA1][16 + khalf * 8];
#pragma unroll
            for (int j = 0; j < 8; ++j) { afrag1[j] = lo[j]; afrag1[8 + j] = hi[j]; }
        }

        acc0 = __builtin_amdgcn_wmma_f32_16x16x32_bf16(
            false, afrag0, false, bfrag, (short)0, acc0, false, false);
        acc1 = __builtin_amdgcn_wmma_f32_16x16x32_bf16(
            false, afrag1, false, bfrag, (short)0, acc1, false, false);

#if USE_ASYNC_LDS
        wait_asynccnt0();   // async DMA into next buffer must land before barrier
#endif
        __syncthreads();
    }

    // Spill z tiles to LDS per C/D layout (VGPR r -> row r or r+8 by lane half).
    {
        const int rb0 = (wave >> 2) * 16 + khalf * 8;
#pragma unroll
        for (int r = 0; r < 8; ++r) lds_z[rb0 + r][gate * 17 + col] = acc0[r];
        const int rb1 = rb0 + 32;
#pragma unroll
        for (int r = 0; r < 8; ++r) lds_z[rb1 + r][gate * 17 + col] = acc1[r];
    }
    __syncthreads();

    // Fused gates + state update: 64 rows x 16 hidden cols, 4 elems/thread.
    const bool last = (t == T_ - 1);
#pragma unroll
    for (int q = 0; q < 4; ++q) {
        const int p = tid + q * 256;
        const int b = p >> 4, n = p & 15;
        const int hglob = g16 + n;
        const float zi = lds_z[b][0 * 17 + n];
        const float zf = lds_z[b][1 * 17 + n];
        const float zg = lds_z[b][2 * 17 + n];
        const float zo = lds_z[b][3 * 17 + n];
        const float ig = sigmoidf_(zi);
        const float fg = sigmoidf_(zf);
        const float gg = tanhf(zg);
        const float og = sigmoidf_(zo);
        const float cold = C[b * H_ + hglob];
        const float cnew = fg * cold + ig * gg;
        const float hnew = og * tanhf(cnew);
        C[b * H_ + hglob] = cnew;
        Aout[b * K_ + E_ + hglob] = (__bf16)hnew;   // h_t for next step's GEMM
        if (last) {
            out[b * H_ + hglob]            = hnew;  // h_T
            out[B_ * H_ + b * H_ + hglob]  = cnew;  // c_T
        }
    }

    // Prefill x_{t+1} slice of Aout (cols blockIdx.x*8 .. +8, all 64 rows).
    if (t + 1 < T_) {
        const int gx8 = blockIdx.x * 8;
#pragma unroll
        for (int q = 0; q < 2; ++q) {
            const int p = tid + q * 256;
            const int b = p >> 3, cc = p & 7;
            Aout[b * K_ + gx8 + cc] = X[((long)b * T_ + (t + 1)) * E_ + gx8 + cc];
        }
    }
}

// ---------------------------------------------------------------------------
// Host launcher. Inputs (reference order): source(i32), emb_table(f32),
// W_ih(f32), W_hh(f32), b(f32). Output: h_T ++ c_T (fp32, 2*B*H floats).
// ---------------------------------------------------------------------------
extern "C" void kernel_launch(void* const* d_in, const int* in_sizes, int n_in,
                              void* d_out, int out_size, void* d_ws, size_t ws_size,
                              hipStream_t stream) {
    const int*   src  = (const int*)d_in[0];
    const float* emb  = (const float*)d_in[1];
    const float* Wih  = (const float*)d_in[2];
    const float* Whh  = (const float*)d_in[3];
    const float* bias = (const float*)d_in[4];
    float* out = (float*)d_out;

    // Workspace carve-out (~46.8 MB), 256B aligned slices.
    char*  ws  = (char*)d_ws;
    size_t off = 0;
    auto carve = [&](size_t bytes) -> void* {
        void* p = ws + off;
        off = (off + bytes + 255) & ~(size_t)255;
        return p;
    };
    __bf16* Wc = (__bf16*)carve((size_t)K_ * N4H * sizeof(__bf16));      // 12.6 MB
    __bf16* X  = (__bf16*)carve((size_t)B_ * T_ * E_ * sizeof(__bf16));  // 33.6 MB
    __bf16* A0 = (__bf16*)carve((size_t)B_ * K_ * sizeof(__bf16));       // 192 KB
    __bf16* A1 = (__bf16*)carve((size_t)B_ * K_ * sizeof(__bf16));       // 192 KB
    float*  C  = (float*) carve((size_t)B_ * H_ * sizeof(float));        // 256 KB

    pack_weights_kernel<<<4096, 256, 0, stream>>>(Wih, Whh, Wc);
    embed_bf16_kernel<<<8192, 256, 0, stream>>>(src, emb, X);
    init_state_kernel<<<(B_ * K_ + 255) / 256, 256, 0, stream>>>(X, A0, C);

    for (int t = 0; t < T_; ++t) {
        const __bf16* Ain  = (t & 1) ? A1 : A0;
        __bf16*       Aout = (t & 1) ? A0 : A1;
        lstm_step_kernel<<<H_ / 16, 256, 0, stream>>>(Ain, Aout, Wc, bias, X,
                                                      C, out, t);
    }
}
